// RelativePosEmb_68607807586791
// MI455X (gfx1250) — compile-verified, matrix operand order checked by reading, not run on hardware
//
#include <hip/hip_runtime.h>

typedef __attribute__((ext_vector_type(16))) _Float16 v16h;
typedef __attribute__((ext_vector_type(8)))  _Float16 v8h;
typedef __attribute__((ext_vector_type(8)))  float    v8f;
typedef __attribute__((ext_vector_type(4)))  float    f4v;

#define LSEQ 1024   // H*W
#define HD   64     // head dim
#define BM   128    // queries per workgroup (8 waves x 16 rows)
#define BN   32     // keys per iteration
#define NWAVE 8
#define NTILE (LSEQ / BN)

// Build a 16-element f16 A/B operand from two contiguous 8-half LDS chunks.
// Element e<8  -> row[8*kh + e] ; element e>=8 -> row[16 + 8*kh + (e-8)]
__device__ __forceinline__ v16h lds_a16(const _Float16* row, int kh) {
  v8h lo = *(const v8h*)(row + 8 * kh);
  v8h hi = *(const v8h*)(row + 16 + 8 * kh);
  v16h r;
#pragma unroll
  for (int i = 0; i < 8; ++i) { r[i] = lo[i]; r[i + 8] = hi[i]; }
  return r;
}

struct StageRegs { v8h kk; v8h vv; };

// Global -> registers: fold position bias into K', convert K'/V to f16.
__device__ __forceinline__ StageRegs stage_load(const float* __restrict__ K,
                                                const float* __restrict__ V,
                                                const float* __restrict__ Hemb,
                                                const float* __restrict__ Wemb,
                                                size_t base, int gk, int sdg) {
  const int kp_ = gk >> 5;   // p (row of key in 32x32 grid)
  const int kw_ = gk & 31;   // q (col of key in 32x32 grid)
  const f4v* kg = (const f4v*)(K    + base + (size_t)gk * HD + sdg);
  const f4v* hg = (const f4v*)(Hemb + kp_ * HD + sdg);
  const f4v* wg = (const f4v*)(Wemb + kw_ * HD + sdg);
  const f4v* vg = (const f4v*)(V    + base + (size_t)gk * HD + sdg);
  f4v ka = kg[0], kb4 = kg[1];
  f4v va = vg[0], vb  = vg[1];
  f4v ha = hg[0], hb  = hg[1];
  f4v wa = wg[0], wb  = wg[1];
  StageRegs st;
#pragma unroll
  for (int i = 0; i < 4; ++i) {
    st.kk[i]     = (_Float16)(ka[i]  + ha[i] + wa[i]);
    st.kk[i + 4] = (_Float16)(kb4[i] + hb[i] + wb[i]);
    st.vv[i]     = (_Float16)va[i];
    st.vv[i + 4] = (_Float16)vb[i];
  }
  return st;
}

// Registers -> LDS: K' key-major, V transposed dim-major.
__device__ __forceinline__ void stage_store(const StageRegs& st,
                                            _Float16* bK, _Float16* bVt,
                                            int skey, int sdg) {
  *(v8h*)(&bK[skey * HD + sdg]) = st.kk;
#pragma unroll
  for (int i = 0; i < 8; ++i) bVt[(sdg + i) * BN + skey] = st.vv[i];
}

__global__ __launch_bounds__(256)
void fa_relpos_kernel(const float* __restrict__ Q,
                      const float* __restrict__ K,
                      const float* __restrict__ V,
                      const float* __restrict__ Wemb,
                      const float* __restrict__ Hemb,
                      float* __restrict__ Out) {
  __shared__ _Float16 sK[2][BN * HD];    // K' tiles (bias folded), key-major
  __shared__ _Float16 sVt[2][HD * BN];   // V tiles transposed, dim-major

  const int pair = blockIdx.y;           // b*8 + n
  const int b    = pair >> 3;
  const int n    = pair & 7;
  const int tid  = threadIdx.x;
  const int lane = tid & 31;
  const int wv   = tid >> 5;             // wave id 0..7
  const int nq   = lane & 15;            // query column / A-row selector
  const int kh   = lane >> 4;            // K-group half selector
  const int qrow = blockIdx.x * BM + wv * 16 + nq;

  const size_t base = (size_t)pair * LSEQ * HD;

  // ---- Q operand (B-matrix layout for S^T = K' x (s*Q)^T), loaded once ----
  // Fold log2(e) into the scale so softmax runs natively in exp2 domain
  // (v_exp_f32 is exp2): softmax(s*logits) == softmax2(s*log2e*logits).
  const float scale = 0.125f * 1.44269504088896340736f;
  const float* qp = Q + base + (size_t)qrow * HD;
  v16h qb0, qb1;
#pragma unroll
  for (int e = 0; e < 8; ++e) {
    qb0[e]     = (_Float16)(qp[      8 * kh + e] * scale);
    qb0[e + 8] = (_Float16)(qp[16 +  8 * kh + e] * scale);
    qb1[e]     = (_Float16)(qp[32 +  8 * kh + e] * scale);
    qb1[e + 8] = (_Float16)(qp[48 +  8 * kh + e] * scale);
  }

  float m_run = -3.0e38f;
  float l_run = 0.0f;
  v8f o0 = {}, o1 = {}, o2 = {}, o3 = {};
  const v8f zero = {};

  const int skey = tid >> 3;             // staging: key 0..31
  const int sdg  = (tid & 7) * 8;        // staging: dim group 0,8,...,56

  // ---- prologue: stage tile 0 into buffer 0 ----
  {
    StageRegs st = stage_load(K, V, Hemb, Wemb, base, skey, sdg);
    stage_store(st, sK[0], sVt[0], skey, sdg);
  }
  __syncthreads();

  for (int it = 0; it < NTILE; ++it) {
    const int cur = it & 1;
    // ---- prefetch next tile into registers (overlaps with WMMA below) ----
    StageRegs st;
    if (it < NTILE - 1)
      st = stage_load(K, V, Hemb, Wemb, base, (it + 1) * BN + skey, sdg);

    // ---- S^T = K'(32x64) x Qs^T(64x16) ----
    const _Float16* arow0 = &sK[cur][nq * HD];
    const _Float16* arow1 = &sK[cur][(nq + 16) * HD];
    v16h a00 = lds_a16(arow0,      kh);
    v16h a01 = lds_a16(arow0 + 32, kh);
    v16h a10 = lds_a16(arow1,      kh);
    v16h a11 = lds_a16(arow1 + 32, kh);
    v8f s0 = __builtin_amdgcn_wmma_f32_16x16x32_f16(false, a00, false, qb0, (short)0, zero, false, false);
    s0     = __builtin_amdgcn_wmma_f32_16x16x32_f16(false, a01, false, qb1, (short)0, s0,   false, false);
    v8f s1 = __builtin_amdgcn_wmma_f32_16x16x32_f16(false, a10, false, qb0, (short)0, zero, false, false);
    s1     = __builtin_amdgcn_wmma_f32_16x16x32_f16(false, a11, false, qb1, (short)0, s1,   false, false);

    // ---- V^T operands: independent of softmax, load early under VALU ----
    v16h va0 = lds_a16(&sVt[cur][(nq +  0) * BN], kh);
    v16h va1 = lds_a16(&sVt[cur][(nq + 16) * BN], kh);
    v16h va2 = lds_a16(&sVt[cur][(nq + 32) * BN], kh);
    v16h va3 = lds_a16(&sVt[cur][(nq + 48) * BN], kh);

    // ---- online softmax in exp2 domain (lane owns query column nq) ----
    float mx = m_run;
#pragma unroll
    for (int r = 0; r < 8; ++r) mx = fmaxf(mx, fmaxf(s0[r], s1[r]));
    mx = fmaxf(mx, __shfl_xor(mx, 16, 32));
    const float alpha = __builtin_amdgcn_exp2f(m_run - mx);
    m_run = mx;

    float rs = 0.0f;
    v16h pb;                             // P^T packed directly into B-operand layout
#pragma unroll
    for (int r = 0; r < 8; ++r) {
      float p0 = __builtin_amdgcn_exp2f(s0[r] - mx);  // key = kb + 8*kh + r
      float p1 = __builtin_amdgcn_exp2f(s1[r] - mx);  // key = kb + 16 + 8*kh + r
      rs += p0 + p1;
      pb[r]     = (_Float16)p0;
      pb[r + 8] = (_Float16)p1;
    }
    rs += __shfl_xor(rs, 16, 32);
    l_run = l_run * alpha + rs;

#pragma unroll
    for (int r = 0; r < 8; ++r) {
      o0[r] *= alpha; o1[r] *= alpha; o2[r] *= alpha; o3[r] *= alpha;
    }

    // ---- O^T += V^T(16x32 per d-tile) x P^T(32x16) ----
    o0 = __builtin_amdgcn_wmma_f32_16x16x32_f16(false, va0, false, pb, (short)0, o0, false, false);
    o1 = __builtin_amdgcn_wmma_f32_16x16x32_f16(false, va1, false, pb, (short)0, o1, false, false);
    o2 = __builtin_amdgcn_wmma_f32_16x16x32_f16(false, va2, false, pb, (short)0, o2, false, false);
    o3 = __builtin_amdgcn_wmma_f32_16x16x32_f16(false, va3, false, pb, (short)0, o3, false, false);

    // ---- soft scheduling hints: batch DS reads ahead of each WMMA burst ----
    __builtin_amdgcn_sched_group_barrier(0x100, 8, 0);   // DS reads: S^T operands
    __builtin_amdgcn_sched_group_barrier(0x008, 4, 0);   // WMMA: S^T
    __builtin_amdgcn_sched_group_barrier(0x100, 8, 0);   // DS reads: V^T operands
    __builtin_amdgcn_sched_group_barrier(0x402, 160, 0); // VALU+TRANS: softmax
    __builtin_amdgcn_sched_group_barrier(0x008, 4, 0);   // WMMA: PV

    // ---- write next tile to the alternate buffer; one barrier per iter ----
    if (it < NTILE - 1) {
      stage_store(st, sK[cur ^ 1], sVt[cur ^ 1], skey, sdg);
      __syncthreads();
    }
  }

  // ---- epilogue: out[b, qh, qw, n*64 + d], non-temporal (write-once) ----
  const float inv = 1.0f / l_run;
  const int qh = qrow >> 5;
  const int qw = qrow & 31;
  float* op = Out + ((((size_t)b * 32 + qh) * 32 + qw) * 512) + n * 64 + 8 * kh;
#define EPI(OACC, TT)                                                            \
  {                                                                              \
    f4v lo4 = { OACC[0] * inv, OACC[1] * inv, OACC[2] * inv, OACC[3] * inv };    \
    f4v hi4 = { OACC[4] * inv, OACC[5] * inv, OACC[6] * inv, OACC[7] * inv };    \
    __builtin_nontemporal_store(lo4, (f4v*)(op + 16 * TT));                      \
    __builtin_nontemporal_store(hi4, (f4v*)(op + 16 * TT + 4));                  \
  }
  EPI(o0, 0)
  EPI(o1, 1)
  EPI(o2, 2)
  EPI(o3, 3)
#undef EPI
}

extern "C" void kernel_launch(void* const* d_in, const int* in_sizes, int n_in,
                              void* d_out, int out_size, void* d_ws, size_t ws_size,
                              hipStream_t stream) {
  const float* q    = (const float*)d_in[0];
  const float* k    = (const float*)d_in[1];
  const float* v    = (const float*)d_in[2];
  const float* wemb = (const float*)d_in[3];
  const float* hemb = (const float*)d_in[4];
  float* out = (float*)d_out;

  dim3 grid(LSEQ / BM, 64);   // (query tiles, b*heads)
  dim3 block(32 * NWAVE);
  fa_relpos_kernel<<<grid, block, 0, stream>>>(q, k, v, wemb, hemb, out);
}
